// SparseMoELayer_90151363543481
// MI455X (gfx1250) — matrix operand order repaired and checked by reading, not compiled, not run on hardware
//
#include <hip/hip_runtime.h>
#include <hip/hip_bf16.h>
#include <cstdint>

// -------- CDNA5 WMMA types --------
typedef __bf16 bf16_t;
typedef __attribute__((ext_vector_type(16))) __bf16 v16bf;
typedef __attribute__((ext_vector_type(8)))  float  v8f;

// ---- fp32 -> bf16 round-to-nearest-even ----
__device__ __forceinline__ uint16_t f2bf_bits(float f) {
    union { float f; uint32_t u; } v; v.f = f;
    uint32_t r = (v.u + 0x7fffu + ((v.u >> 16) & 1u)) >> 16;
    return (uint16_t)r;
}
__device__ __forceinline__ bf16_t f2bf(float f) {
    union { uint16_t s; bf16_t b; } o; o.s = f2bf_bits(f);
    return o.b;
}

// =====================================================================
// 1a) elementwise fp32 -> bf16 (x stays K-major)
// =====================================================================
__global__ __launch_bounds__(256) void f32_to_bf16_kernel(
    const float* __restrict__ src, bf16_t* __restrict__ dst, long n4) {
    long i = (long)blockIdx.x * blockDim.x + threadIdx.x;
    if (i >= n4) return;
    float4 v = ((const float4*)src)[i];
    union { uint16_t s[4]; uint2 u; } o;
    o.s[0] = f2bf_bits(v.x); o.s[1] = f2bf_bits(v.y);
    o.s[2] = f2bf_bits(v.z); o.s[3] = f2bf_bits(v.w);
    ((uint2*)dst)[i] = o.u;
}

// =====================================================================
// 1b) fp32 [E][K][N] -> bf16 [E][N][K] transpose (32x32 tiles via LDS)
// =====================================================================
__global__ __launch_bounds__(256) void transpose_convert_kernel(
    const float* __restrict__ src, bf16_t* __restrict__ dst, int K, int N) {
    const int e  = blockIdx.z;
    const int k0 = blockIdx.y * 32;
    const int n0 = blockIdx.x * 32;
    __shared__ bf16_t tile[32][33];
    const int tid = threadIdx.x;
    const int r  = tid >> 3;        // 0..31
    const int c4 = (tid & 7) * 4;   // 0..28
    const float* sp = src + (size_t)e * K * N + (size_t)(k0 + r) * N + n0 + c4;
    float4 v = *(const float4*)sp;
    tile[r][c4 + 0] = f2bf(v.x);
    tile[r][c4 + 1] = f2bf(v.y);
    tile[r][c4 + 2] = f2bf(v.z);
    tile[r][c4 + 3] = f2bf(v.w);
    __syncthreads();
    union { bf16_t b[4]; uint2 u; } o;
    o.b[0] = tile[c4 + 0][r];
    o.b[1] = tile[c4 + 1][r];
    o.b[2] = tile[c4 + 2][r];
    o.b[3] = tile[c4 + 3][r];
    bf16_t* dp = dst + (size_t)e * N * K + (size_t)(n0 + r) * K + k0 + c4;
    *(uint2*)dp = o.u;
}

// =====================================================================
// 2) Router: one wave32 per token. 8 logits, top-2, softmax-of-2.
// =====================================================================
__global__ __launch_bounds__(256) void router_kernel(
    const float* __restrict__ x, const float* __restrict__ rw,
    const float* __restrict__ rb,
    int* __restrict__ expert_id, float* __restrict__ gate,
    int* __restrict__ counts, int T, int H) {
    const int lane = threadIdx.x & 31;
    const int t = blockIdx.x * 8 + (threadIdx.x >> 5);
    if (t >= T) return;
    const float* xr = x + (size_t)t * H;
    float acc[8];
#pragma unroll
    for (int e = 0; e < 8; ++e) acc[e] = 0.f;
    for (int h = lane; h < H; h += 32) {
        float xv = xr[h];
        const float* w = rw + (size_t)h * 8;
#pragma unroll
        for (int e = 0; e < 8; ++e) acc[e] += xv * w[e];
    }
#pragma unroll
    for (int off = 16; off > 0; off >>= 1) {
#pragma unroll
        for (int e = 0; e < 8; ++e) acc[e] += __shfl_xor(acc[e], off, 32);
    }
    if (lane == 0) {
        float v0 = acc[0] + rb[0];
        float best = v0, second = -3.402823466e+38f;
        int bi = 0, si = 1;
#pragma unroll
        for (int e = 1; e < 8; ++e) {
            float v = acc[e] + rb[e];
            if (v > best)        { second = best; si = bi; best = v; bi = e; }
            else if (v > second) { second = v; si = e; }
        }
        float p0 = 1.f / (1.f + expf(second - best));
        float p1 = 1.f - p0;
        expert_id[2 * t]     = bi; gate[2 * t]     = p0;
        expert_id[2 * t + 1] = si; gate[2 * t + 1] = p1;
        atomicAdd(&counts[bi], 1);
        atomicAdd(&counts[si], 1);
    }
}

// =====================================================================
// 3a) tiny exclusive scan of expert counts
// =====================================================================
__global__ void scan_kernel(const int* __restrict__ counts, int* __restrict__ offsets) {
    if (threadIdx.x == 0 && blockIdx.x == 0) {
        int s = 0;
#pragma unroll
        for (int e = 0; e < 8; ++e) { offsets[e] = s; s += counts[e]; }
        offsets[8] = s;
    }
}

// =====================================================================
// 3b) deterministic stable compaction: one block per expert (ballot scan)
// =====================================================================
__global__ __launch_bounds__(1024) void sort_kernel(
    const int* __restrict__ expert_id, const int* __restrict__ offsets,
    int* __restrict__ rows_token, int* __restrict__ row_of_assign, int nA) {
    const int e = blockIdx.x;
    __shared__ int wcnt[32];
    __shared__ int wpref[32];
    __shared__ int s_base;
    const int tid = threadIdx.x, lane = tid & 31, w = tid >> 5;
    if (tid == 0) s_base = offsets[e];
    __syncthreads();
    for (int c0 = 0; c0 < nA; c0 += 1024) {
        int a = c0 + tid;
        bool flag = (a < nA) && (expert_id[a] == e);
        unsigned long long mask = __ballot(flag);
        int within = __popcll(mask & ((1ull << lane) - 1ull));
        if (lane == 0) wcnt[w] = __popcll(mask);
        __syncthreads();
        if (tid == 0) {
            int s = s_base;
#pragma unroll
            for (int i = 0; i < 32; ++i) { wpref[i] = s; s += wcnt[i]; }
            s_base = s;
        }
        __syncthreads();
        if (flag) {
            int pos = wpref[w] + within;
            rows_token[pos] = a >> 1;
            row_of_assign[a] = pos;
        }
    }
}

// =====================================================================
// 4) Grouped GEMM, V_WMMA_F32_16X16X32_BF16 + GLOBAL_LOAD_ASYNC_TO_LDS.
//    Block: 256 thr (8 waves), tile 64(M) x 128(N), BK=32.
//    Each wave: 32x32 C block = 4 WMMA / K-step, fragments via ds_load_b128.
//    Double-buffered LDS filled by async memory->LDS DMA (ASYNCcnt), one
//    barrier per K-step. Branch-free steady-state loop: the last K-step is
//    peeled and the double-buffer toggle is an XOR of the two LDS offsets.
// =====================================================================
template <bool GATHER, bool RELU_BF16>
__global__ __launch_bounds__(256) void moe_gemm_kernel(
    const bf16_t* __restrict__ A,      // [rows, K] (x_bf or h1), K-major
    const bf16_t* __restrict__ BwT,    // [E, N, K] bf16 transposed weights
    const float*  __restrict__ bias,   // [E, N]
    void*         __restrict__ Cout,   // [2T, N] bf16 (h1) or f32 (eout)
    const int*    __restrict__ rows_token,
    const int*    __restrict__ offsets,
    int K, int N) {
    const int e   = blockIdx.z;
    const int off = offsets[e];
    const int cnt = offsets[e + 1] - off;
    const int m_tile = blockIdx.y * 64;
    if (m_tile >= cnt) return;               // uniform exit, EXEC stays full
    const int n0 = blockIdx.x * 128;

    __shared__ bf16_t As[2][64][48];         // 96B row stride (16B-aligned frags)
    __shared__ bf16_t Bs[2][128][48];        // N-major: [n][k]

    const int tid  = threadIdx.x;
    const int lane = tid & 31;
    const int wave = tid >> 5;
    const int wm = (wave >> 2) * 32;         // wave M offset: 0 / 32
    const int wn = (wave & 3) * 32;          // wave N offset: 0/32/64/96

    // A staging: 64 rows x 4 thr, 16B each, async -> LDS
    const int arow = tid >> 2;
    const int acol = (tid & 3) * 8;
    int gr = m_tile + arow;
    if (gr >= cnt) gr = cnt - 1;             // clamp: row masked at epilogue
    const int rowidx = GATHER ? rows_token[off + gr] : (off + gr);
    const bf16_t* aglob = A + (size_t)rowidx * K + acol;

    // B staging: 128 n-rows x 2 thr, 32B each (2 async b128)
    const int brow = tid >> 1;
    const int bk   = (tid & 1) * 16;
    const bf16_t* bglob = BwT + (size_t)e * N * K + (size_t)(n0 + brow) * K + bk;

    const uint32_t aoff0 = (uint32_t)(uintptr_t)&As[0][arow][acol];
    const uint32_t aoff1 = (uint32_t)(uintptr_t)&As[1][arow][acol];
    const uint32_t boff0 = (uint32_t)(uintptr_t)&Bs[0][brow][bk];
    const uint32_t boff1 = (uint32_t)(uintptr_t)&Bs[1][brow][bk];
    const uint32_t axor = aoff0 ^ aoff1;
    const uint32_t bxor = boff0 ^ boff1;

    v8f c00 = {0.f,0.f,0.f,0.f,0.f,0.f,0.f,0.f};
    v8f c01 = {0.f,0.f,0.f,0.f,0.f,0.f,0.f,0.f};
    v8f c10 = {0.f,0.f,0.f,0.f,0.f,0.f,0.f,0.f};
    v8f c11 = {0.f,0.f,0.f,0.f,0.f,0.f,0.f,0.f};

    const int lh = lane & 15;
    const int kA = (lane < 16) ? 0 : 8;      // A frag: K kA..kA+7, kA+16..kA+23
    const int kB = (lane < 16) ? 0 : 8;      // B frag: K kB..kB+7, kB+16..kB+23

    // issue async DMA loads for K-slice k into LDS offsets (ao, bo)
    auto issue = [&](uint32_t ao, uint32_t bo, int k) {
        const bf16_t* ag = aglob + k;
        const bf16_t* bg = bglob + k;
        asm volatile("global_load_async_to_lds_b128 %0, %1, off"
                     :: "v"(ao), "v"(ag) : "memory");
        asm volatile("global_load_async_to_lds_b128 %0, %1, off"
                     :: "v"(bo), "v"(bg) : "memory");
        // offset applies to BOTH the global and LDS address (ISA 8.2)
        asm volatile("global_load_async_to_lds_b128 %0, %1, off offset:16"
                     :: "v"(bo), "v"(bg) : "memory");
    };

    // fragments + 4 WMMA from buffer b
    auto compute = [&](int b) {
        union FB { uint4 u[2]; v16bf v; };
        FB a0, a1, b0, b1;
        const bf16_t* ap0 = &As[b][wm + lh][kA];
        const bf16_t* ap1 = &As[b][wm + 16 + lh][kA];
        a0.u[0] = *(const uint4*)(ap0);       a0.u[1] = *(const uint4*)(ap0 + 16);
        a1.u[0] = *(const uint4*)(ap1);       a1.u[1] = *(const uint4*)(ap1 + 16);
        const bf16_t* bp0 = &Bs[b][wn + lh][kB];
        const bf16_t* bp1 = &Bs[b][wn + 16 + lh][kB];
        b0.u[0] = *(const uint4*)(bp0);       b0.u[1] = *(const uint4*)(bp0 + 16);
        b1.u[0] = *(const uint4*)(bp1);       b1.u[1] = *(const uint4*)(bp1 + 16);
        c00 = __builtin_amdgcn_wmma_f32_16x16x32_bf16(false, a0.v, false, b0.v, (short)0, c00, false, false);
        c01 = __builtin_amdgcn_wmma_f32_16x16x32_bf16(false, a0.v, false, b1.v, (short)0, c01, false, false);
        c10 = __builtin_amdgcn_wmma_f32_16x16x32_bf16(false, a1.v, false, b0.v, (short)0, c10, false, false);
        c11 = __builtin_amdgcn_wmma_f32_16x16x32_bf16(false, a1.v, false, b1.v, (short)0, c11, false, false);
    };

    uint32_t ao = aoff0, bo = boff0;
    issue(ao, bo, 0);
    int buf = 0;
    int k0 = 0;
    // steady state: branch-free body (K >= 64 always here)
    for (; k0 < K - 32; k0 += 32) {
        asm volatile("s_wait_asynccnt 0x0" ::: "memory");   // my DMAs landed
        __syncthreads();                                    // everyone's landed
        ao ^= axor; bo ^= bxor;                             // toggle buffer
        issue(ao, bo, k0 + 32);
        compute(buf);
        buf ^= 1;
    }
    // peeled last K-step: no further issue
    asm volatile("s_wait_asynccnt 0x0" ::: "memory");
    __syncthreads();
    compute(buf);

    // ---- epilogue: C VGPR i -> row base+i (lanes>=16: +8), col = lane&15 ----
    const int mb   = m_tile + wm + ((lane < 16) ? 0 : 8);
    const int colA = n0 + wn + lh;
    const int colB = colA + 16;
    const float biasA = bias[(size_t)e * N + colA];
    const float biasB = bias[(size_t)e * N + colB];
#pragma unroll
    for (int i = 0; i < 8; ++i) {
        int m0 = mb + i;          // c00 / c01
        int m1 = mb + 16 + i;     // c10 / c11
        if (m0 < cnt) {
            size_t r = (size_t)(off + m0);
            float v0 = c00[i] + biasA;
            float v1 = c01[i] + biasB;
            if (RELU_BF16) {
                bf16_t* C = (bf16_t*)Cout;
                C[r * (size_t)N + colA] = f2bf(v0 > 0.f ? v0 : 0.f);
                C[r * (size_t)N + colB] = f2bf(v1 > 0.f ? v1 : 0.f);
            } else {
                float* C = (float*)Cout;
                C[r * (size_t)N + colA] = v0;
                C[r * (size_t)N + colB] = v1;
            }
        }
        if (m1 < cnt) {
            size_t r = (size_t)(off + m1);
            float v0 = c10[i] + biasA;
            float v1 = c11[i] + biasB;
            if (RELU_BF16) {
                bf16_t* C = (bf16_t*)Cout;
                C[r * (size_t)N + colA] = f2bf(v0 > 0.f ? v0 : 0.f);
                C[r * (size_t)N + colB] = f2bf(v1 > 0.f ? v1 : 0.f);
            } else {
                float* C = (float*)Cout;
                C[r * (size_t)N + colA] = v0;
                C[r * (size_t)N + colB] = v1;
            }
        }
    }
}

// =====================================================================
// 5) deterministic combine: out[t] = g0*eout[r0] + g1*eout[r1]
// =====================================================================
__global__ __launch_bounds__(256) void combine_kernel(
    const float* __restrict__ eout, const int* __restrict__ roa,
    const float* __restrict__ gate, float* __restrict__ out, int H) {
    const int t = blockIdx.x;
    const int r0 = roa[2 * t], r1 = roa[2 * t + 1];
    const float g0 = gate[2 * t], g1 = gate[2 * t + 1];
    const float4* p0 = (const float4*)(eout + (size_t)r0 * H);
    const float4* p1 = (const float4*)(eout + (size_t)r1 * H);
    float4* po = (float4*)(out + (size_t)t * H);
    for (int i = threadIdx.x; i < (H >> 2); i += blockDim.x) {
        float4 a = p0[i], b = p1[i];
        float4 r;
        r.x = g0 * a.x + g1 * b.x;
        r.y = g0 * a.y + g1 * b.y;
        r.z = g0 * a.z + g1 * b.z;
        r.w = g0 * a.w + g1 * b.w;
        po[i] = r;
    }
}

// =====================================================================
// host-side launch
// =====================================================================
extern "C" void kernel_launch(void* const* d_in, const int* in_sizes, int n_in,
                              void* d_out, int out_size, void* d_ws, size_t ws_size,
                              hipStream_t stream) {
    const float* x  = (const float*)d_in[0];
    const float* rw = (const float*)d_in[1];
    const float* rb = (const float*)d_in[2];
    const float* w1 = (const float*)d_in[3];
    const float* b1 = (const float*)d_in[4];
    const float* w2 = (const float*)d_in[5];
    const float* b2 = (const float*)d_in[6];
    float* out = (float*)d_out;

    const int  E  = in_sizes[2];                 // 8
    const int  H  = in_sizes[1] / E;             // 1024
    const long T  = (long)in_sizes[0] / H;       // 8192
    const int  F  = in_sizes[4] / E;             // 2048
    const long nA = 2 * T;

    uintptr_t p = (uintptr_t)d_ws;
    auto take = [&](size_t bytes) -> void* {
        p = (p + 255) & ~(uintptr_t)255;
        void* q = (void*)p;
        p += bytes;
        return q;
    };
    bf16_t* x_bf   = (bf16_t*)take((size_t)T * H * 2);
    bf16_t* w1t_bf = (bf16_t*)take((size_t)E * H * F * 2);   // [E][F][H]
    bf16_t* w2t_bf = (bf16_t*)take((size_t)E * F * H * 2);   // [E][H][F]
    bf16_t* h1     = (bf16_t*)take((size_t)nA * F * 2);
    float*  eout   = (float*) take((size_t)nA * H * 4);
    int*    eid    = (int*)   take((size_t)nA * 4);
    float*  gatev  = (float*) take((size_t)nA * 4);
    int*    rtok   = (int*)   take((size_t)nA * 4);
    int*    roa    = (int*)   take((size_t)nA * 4);
    int*    counts = (int*)   take(16 * 4);
    int*    offs   = (int*)   take(16 * 4);
    (void)ws_size; (void)n_in; (void)out_size;

    hipMemsetAsync(counts, 0, 16 * 4, stream);

    // 1) conversions: x stays [T][H]; weights transposed to N-major
    { long n4 = (long)T * H / 4;
      f32_to_bf16_kernel<<<dim3((unsigned)((n4 + 255) / 256)), 256, 0, stream>>>(x, x_bf, n4); }
    transpose_convert_kernel<<<dim3((unsigned)(F / 32), (unsigned)(H / 32), (unsigned)E),
                               256, 0, stream>>>(w1, w1t_bf, H, F);   // K=H, N=F
    transpose_convert_kernel<<<dim3((unsigned)(H / 32), (unsigned)(F / 32), (unsigned)E),
                               256, 0, stream>>>(w2, w2t_bf, F, H);   // K=F, N=H

    // 2) router
    router_kernel<<<dim3((unsigned)((T + 7) / 8)), 256, 0, stream>>>(
        x, rw, rb, eid, gatev, counts, (int)T, H);

    // 3) offsets + deterministic compaction
    scan_kernel<<<1, 32, 0, stream>>>(counts, offs);
    sort_kernel<<<E, 1024, 0, stream>>>(eid, offs, rtok, roa, (int)nA);

    // 4) expert GEMMs (WMMA bf16 + async LDS staging)
    dim3 g1((unsigned)(F / 128), (unsigned)(T / 64), (unsigned)E);
    moe_gemm_kernel<true,  true ><<<g1, 256, 0, stream>>>(x_bf, w1t_bf, b1, (void*)h1,  rtok, offs, H, F);
    dim3 g2((unsigned)(H / 128), (unsigned)(T / 64), (unsigned)E);
    moe_gemm_kernel<false, false><<<g2, 256, 0, stream>>>(h1,  w2t_bf, b2, (void*)eout, rtok, offs, F, H);

    // 5) deterministic gather-combine
    combine_kernel<<<dim3((unsigned)T), 256, 0, stream>>>(eout, roa, gatev, out, H);
}
